// MultiFilterConvDotProductAttention_4088808866278
// MI455X (gfx1250) — matrix operand order repaired and checked by reading, not compiled, not run on hardware
//
#include <hip/hip_runtime.h>
#include <hip/hip_bf16.h>
#include <math.h>

// MI455X / gfx1250, wave32. Two-pass pipeline:
//   K1: dp = (Q/8)·K^T   via v_wmma_f32_16x16x32_bf16 with 2-term bf16 split (fp32-accurate)
//   K2: conv3x3(4f)+leaky+linear+mask -> row softmax -> attn·V (WMMA, same split)
// d_ws must hold B*H*S*S floats = 64 MB (dp stays L2-resident: 64MB < 192MB L2).

typedef __attribute__((ext_vector_type(16))) __bf16 v16bf;
typedef __attribute__((ext_vector_type(8)))  float  v8f;

constexpr int S_DIM = 1024;
constexpr int D_DIM = 64;
constexpr int H_DIM = 8;
constexpr int BH    = 16;   // B*H

__device__ __forceinline__ __bf16 top16(float f) {
    unsigned u = __float_as_uint(f);
    unsigned short h = (unsigned short)(u >> 16);
    __bf16 r;
    __builtin_memcpy(&r, &h, 2);
    return r;
}
__device__ __forceinline__ float hi_f32(float f) {
    return __uint_as_float(__float_as_uint(f) & 0xffff0000u);
}

// Split 16 floats into bf16 hi/lo fragments (all indices constant after unroll).
__device__ __forceinline__ void split16(const float* __restrict__ f, v16bf& hi, v16bf& lo) {
#pragma unroll
    for (int j = 0; j < 16; ++j) {
        const float hf = hi_f32(f[j]);
        hi[j] = top16(hf);
        lo[j] = top16(f[j] - hf);
    }
}

// ---------------------------------------------------------------------------
// Kernel 1: dp[bh][m][t] = sum_d (Q[bh][m][d]/8) * K[bh][t][d]
// One wave (32 threads) computes a 16x64 tile of dp.
// A-layout per lane: two contiguous 8-float chunks  -> float4 (b128) loads.
// B-layout per lane: one contiguous 16-float chunk  -> float4 (b128) loads.
// ---------------------------------------------------------------------------
__global__ __launch_bounds__(32) void qk_scores_kernel(
    const float* __restrict__ q, const float* __restrict__ kmat,
    float* __restrict__ dp)
{
    const int lane  = threadIdx.x & 31;
    const int bh    = blockIdx.z;
    const int m0    = blockIdx.y * 16;
    const int n0b   = blockIdx.x * 64;
    const int row   = lane & 15;
    const int base  = (lane < 16) ? 0 : 16;   // A-matrix K-split (contiguous, matches B)
    const int khalf = (lane < 16) ? 0 : 16;   // B-matrix K-split (contiguous, matches A)

    // Build A (Q rows, pre-scaled by 1/sqrt(D)=1/8) for both 32-wide K steps.
    const float4* qv = reinterpret_cast<const float4*>(
        q + ((size_t)bh * S_DIM + (m0 + row)) * D_DIM);
    v16bf ah0, al0, ah1, al1;
#pragma unroll
    for (int ks = 0; ks < 2; ++ks) {
        const int k0 = ks * 32;
        const float4 c0 = qv[(k0 + base) >> 2];
        const float4 c1 = qv[((k0 + base) >> 2) + 1];
        const float4 c2 = qv[(k0 + base + 8) >> 2];
        const float4 c3 = qv[((k0 + base + 8) >> 2) + 1];
        float fa[16] = {c0.x, c0.y, c0.z, c0.w, c1.x, c1.y, c1.z, c1.w,
                        c2.x, c2.y, c2.z, c2.w, c3.x, c3.y, c3.z, c3.w};
#pragma unroll
        for (int j = 0; j < 16; ++j) fa[j] *= 0.125f;
        if (ks == 0) split16(fa, ah0, al0);
        else         split16(fa, ah1, al1);
    }

    float* dpb = dp + (size_t)bh * S_DIM * S_DIM;
#pragma unroll
    for (int nt = 0; nt < 4; ++nt) {
        const int n0 = n0b + nt * 16;
        const float4* kv = reinterpret_cast<const float4*>(
            kmat + ((size_t)bh * S_DIM + (n0 + row)) * D_DIM);
        v8f c = {0.f, 0.f, 0.f, 0.f, 0.f, 0.f, 0.f, 0.f};
#pragma unroll
        for (int ks = 0; ks < 2; ++ks) {
            const int k0 = ks * 32;
            const float4 b0 = kv[(k0 + khalf) >> 2];
            const float4 b1 = kv[((k0 + khalf) >> 2) + 1];
            const float4 b2 = kv[((k0 + khalf) >> 2) + 2];
            const float4 b3 = kv[((k0 + khalf) >> 2) + 3];
            const float fb[16] = {b0.x, b0.y, b0.z, b0.w, b1.x, b1.y, b1.z, b1.w,
                                  b2.x, b2.y, b2.z, b2.w, b3.x, b3.y, b3.z, b3.w};
            v16bf bhv, blv;
            split16(fb, bhv, blv);
            const v16bf& ah = (ks == 0) ? ah0 : ah1;
            const v16bf& al = (ks == 0) ? al0 : al1;
            // hi*hi + hi*lo + lo*hi  (fp32-accurate bf16 emulation)
            c = __builtin_amdgcn_wmma_f32_16x16x32_bf16(false, ah, false, bhv, (short)0, c, false, false);
            c = __builtin_amdgcn_wmma_f32_16x16x32_bf16(false, ah, false, blv, (short)0, c, false, false);
            c = __builtin_amdgcn_wmma_f32_16x16x32_bf16(false, al, false, bhv, (short)0, c, false, false);
        }
        const int t = n0 + row;
#pragma unroll
        for (int r = 0; r < 8; ++r) {
            const int m = m0 + r + ((lane < 16) ? 0 : 8);
            dpb[(size_t)m * S_DIM + t] = c[r];
        }
    }
}

// ---------------------------------------------------------------------------
// Kernel 2: per (bh, 16-row block): conv+leaky+linear+mask -> softmax -> attn·V
// 256 threads (8 waves). LDS holds preattn/attn for 16 rows (exactly 64 KB).
// ---------------------------------------------------------------------------
__global__ __launch_bounds__(256) void conv_softmax_av_kernel(
    const float* __restrict__ dp, const float* __restrict__ vals,
    const int* __restrict__ mask,
    const float* __restrict__ conv_w, const float* __restrict__ conv_b,
    const float* __restrict__ lin_w, const float* __restrict__ lin_b,
    float* __restrict__ out)
{
    __shared__ float pre[16 * S_DIM];   // 65536 bytes
    const int tid = threadIdx.x;
    const int bh  = blockIdx.y;
    const int b   = bh / H_DIM;
    const int s0  = blockIdx.x * 16;
    const float* dpb = dp + (size_t)bh * S_DIM * S_DIM;

    // small weights into registers (L2-cached reads)
    float w[36];
#pragma unroll
    for (int i = 0; i < 36; ++i) w[i] = conv_w[i];
    float cbl[4], lw[4];
#pragma unroll
    for (int f = 0; f < 4; ++f) { cbl[f] = conv_b[f]; lw[f] = lin_w[f]; }
    const float lb = lin_b[0];

    // ---- Phase A: preattn = lin( leaky( conv3x3(dp) ) ) with mask ----
    for (int e = tid; e < 16 * S_DIM; e += 256) {
        const int r   = e >> 10;
        const int col = e & (S_DIM - 1);
        const int s   = s0 + r;
        float a0 = cbl[0], a1 = cbl[1], a2 = cbl[2], a3 = cbl[3];
#pragma unroll
        for (int di = -1; di <= 1; ++di) {
            const int sr = s + di;
            if ((unsigned)sr >= (unsigned)S_DIM) continue;
            const float* rowp = dpb + (size_t)sr * S_DIM;
#pragma unroll
            for (int dj = -1; dj <= 1; ++dj) {
                const int cc = col + dj;
                if ((unsigned)cc >= (unsigned)S_DIM) continue;
                const float d  = rowp[cc];
                const int   wi = (di + 1) * 3 + (dj + 1);
                a0 += w[0 * 9 + wi] * d;
                a1 += w[1 * 9 + wi] * d;
                a2 += w[2 * 9 + wi] * d;
                a3 += w[3 * 9 + wi] * d;
            }
        }
        a0 = (a0 > 0.f) ? a0 : 0.01f * a0;
        a1 = (a1 > 0.f) ? a1 : 0.01f * a1;
        a2 = (a2 > 0.f) ? a2 : 0.01f * a2;
        a3 = (a3 > 0.f) ? a3 : 0.01f * a3;
        float p = lb + lw[0] * a0 + lw[1] * a1 + lw[2] * a2 + lw[3] * a3;
        const int mv = mask[(size_t)b * S_DIM * S_DIM + (size_t)s * S_DIM + col];
        pre[r * S_DIM + col] = (mv == 0) ? -1.0e30f : p;
    }
    __syncthreads();

    // ---- Phase B: row softmax (2 rows per wave, wave32 butterflies) ----
    const int lane = tid & 31;
    const int wid  = tid >> 5;
    for (int rr = 0; rr < 2; ++rr) {
        const int r = wid * 2 + rr;
        float* rowp = pre + r * S_DIM;
        float m = -3.0e38f;
#pragma unroll 4
        for (int i = 0; i < 32; ++i) m = fmaxf(m, rowp[lane + (i << 5)]);
        for (int off = 16; off; off >>= 1) m = fmaxf(m, __shfl_xor(m, off, 32));
        float ssum = 0.f;
#pragma unroll 4
        for (int i = 0; i < 32; ++i) {
            const int idx = lane + (i << 5);
            const float ev = __expf(rowp[idx] - m);
            rowp[idx] = ev;
            ssum += ev;
        }
        for (int off = 16; off; off >>= 1) ssum += __shfl_xor(ssum, off, 32);
        const float inv = 1.0f / ssum;
#pragma unroll 4
        for (int i = 0; i < 32; ++i) rowp[lane + (i << 5)] *= inv;
    }
    __syncthreads();

    // ---- Phase C: out[16x64] = attn[16x1024] · V[1024x64] via WMMA ----
    if (wid < 4) {
        const int n0    = wid * 16;
        const int row   = lane & 15;
        const int base  = (lane < 16) ? 0 : 16;
        const int khalf = (lane < 16) ? 0 : 16;
        const float* vb = vals + (size_t)bh * S_DIM * D_DIM;
        // attn row base: row*S_DIM floats = 4KB aligned; chunks 16B aligned.
        const float4* arow = reinterpret_cast<const float4*>(pre + row * S_DIM);
        v8f c = {0.f, 0.f, 0.f, 0.f, 0.f, 0.f, 0.f, 0.f};
        for (int k0 = 0; k0 < S_DIM; k0 += 32) {
            const float4 a0 = arow[(k0 + base) >> 2];
            const float4 a1 = arow[((k0 + base) >> 2) + 1];
            const float4 a2 = arow[(k0 + base + 8) >> 2];
            const float4 a3 = arow[((k0 + base + 8) >> 2) + 1];
            const float fa[16] = {a0.x, a0.y, a0.z, a0.w, a1.x, a1.y, a1.z, a1.w,
                                  a2.x, a2.y, a2.z, a2.w, a3.x, a3.y, a3.z, a3.w};
            v16bf ahv, alv;
            split16(fa, ahv, alv);

            float fb[16];
#pragma unroll
            for (int j = 0; j < 16; ++j)
                fb[j] = vb[(size_t)(k0 + khalf + j) * D_DIM + (n0 + row)];
            v16bf bhv, blv;
            split16(fb, bhv, blv);

            c = __builtin_amdgcn_wmma_f32_16x16x32_bf16(false, ahv, false, bhv, (short)0, c, false, false);
            c = __builtin_amdgcn_wmma_f32_16x16x32_bf16(false, ahv, false, blv, (short)0, c, false, false);
            c = __builtin_amdgcn_wmma_f32_16x16x32_bf16(false, alv, false, bhv, (short)0, c, false, false);
        }
#pragma unroll
        for (int r8 = 0; r8 < 8; ++r8) {
            const int m = r8 + ((lane < 16) ? 0 : 8);
            out[(size_t)bh * S_DIM * D_DIM + (size_t)(s0 + m) * D_DIM + (n0 + row)] = c[r8];
        }
    }
}

// ---------------------------------------------------------------------------
extern "C" void kernel_launch(void* const* d_in, const int* in_sizes, int n_in,
                              void* d_out, int out_size, void* d_ws, size_t ws_size,
                              hipStream_t stream) {
    (void)in_sizes; (void)n_in; (void)out_size; (void)ws_size;
    const float* q      = (const float*)d_in[0];
    const float* k      = (const float*)d_in[1];
    const float* v      = (const float*)d_in[2];
    const int*   mask   = (const int*)  d_in[3];
    const float* conv_w = (const float*)d_in[4];
    const float* conv_b = (const float*)d_in[5];
    const float* lin_w  = (const float*)d_in[6];
    const float* lin_b  = (const float*)d_in[7];
    float* out = (float*)d_out;
    float* dp  = (float*)d_ws;   // requires BH*S*S*4 = 64 MB scratch

    dim3 g1(S_DIM / 64, S_DIM / 16, BH);     // 16 x 64 x 16 waves of 32
    qk_scores_kernel<<<g1, dim3(32), 0, stream>>>(q, k, dp);

    dim3 g2(S_DIM / 16, BH);                 // 64 x 16 blocks of 256
    conv_softmax_av_kernel<<<g2, dim3(256), 0, stream>>>(
        dp, v, mask, conv_w, conv_b, lin_w, lin_b, out);
}
